// Attention_7275674600158
// MI455X (gfx1250) — compile-verified, matrix operand order checked by reading, not run on hardware
//
#include <hip/hip_runtime.h>

// ---------------- problem constants ----------------
#define SB   2      // batch
#define SS   2048   // seq len
#define SD   1024   // model dim
#define SH   16     // q heads
#define SHKV 4      // kv heads
#define SHD  64     // head dim
#define SKV  (SHKV * SHD)   // 256
#define SM   (SB * SS)      // 4096 tokens

typedef __bf16 bf16_t;
typedef __attribute__((ext_vector_type(16))) __bf16 bf16x16;
typedef __attribute__((ext_vector_type(8)))  __bf16 bf16x8;
typedef __attribute__((ext_vector_type(4)))  __bf16 bf16x4;
typedef __attribute__((ext_vector_type(8)))  float  f32x8;

// D(f32 16x16) = A(bf16 16x32) * B(bf16 32x16) + C
__device__ __forceinline__ f32x8 wmma_bf16f32(bf16x16 a, bf16x16 b, f32x8 c) {
  return __builtin_amdgcn_wmma_f32_16x16x32_bf16(
      /*neg_a=*/false, a, /*neg_b=*/false, b,
      /*c_mod=*/(short)0, c, /*reuse_a=*/false, /*reuse_b=*/false);
}

// Fragment loader for A (row = M per lane) or B-from-transposed (row = N per
// lane). Wave32 16-bit layout: lanes 0-15 hold k = [0..8)+[16..24),
// lanes 16-31 hold k = [8..16)+[24..32) -> two contiguous 16B runs.
__device__ __forceinline__ bf16x16 load_frag(const bf16_t* base, int ld, int row, int k0) {
  const int half = (threadIdx.x >> 4) & 1;
  const bf16_t* p = base + row * ld + k0 + half * 8;
  bf16x8 lo = *reinterpret_cast<const bf16x8*>(p);
  bf16x8 hi = *reinterpret_cast<const bf16x8*>(p + 16);
  bf16x16 f;
#pragma unroll
  for (int i = 0; i < 8; ++i) { f[i] = lo[i]; f[i + 8] = hi[i]; }
  return f;
}

// ---------------- f32 -> bf16 conversion ----------------
__global__ void cvt_bf16_kernel(const float* __restrict__ in, bf16_t* __restrict__ out, int n) {
  int i = (blockIdx.x * 256 + threadIdx.x) * 4;
  if (i + 3 < n) {
#pragma unroll
    for (int j = 0; j < 4; ++j) out[i + j] = (bf16_t)in[i + j];
  }
}

// ---------------- LayerNorm (per token, D=1024) ----------------
__global__ __launch_bounds__(256) void layernorm_kernel(
    const float* __restrict__ x, const float* __restrict__ w,
    const float* __restrict__ b, bf16_t* __restrict__ out) {
  __shared__ float red[256];
  const int t = threadIdx.x;
  const float* row = x + (size_t)blockIdx.x * SD;
  float xv[4], s = 0.f;
#pragma unroll
  for (int i = 0; i < 4; ++i) { xv[i] = row[t * 4 + i]; s += xv[i]; }
  red[t] = s; __syncthreads();
  for (int o = 128; o > 0; o >>= 1) { if (t < o) red[t] += red[t + o]; __syncthreads(); }
  const float mu = red[0] * (1.0f / SD);
  __syncthreads();
  float sq = 0.f;
#pragma unroll
  for (int i = 0; i < 4; ++i) { float d = xv[i] - mu; sq += d * d; }
  red[t] = sq; __syncthreads();
  for (int o = 128; o > 0; o >>= 1) { if (t < o) red[t] += red[t + o]; __syncthreads(); }
  const float r = rsqrtf(red[0] * (1.0f / SD) + 1e-5f);
  bf16_t* orow = out + (size_t)blockIdx.x * SD;
#pragma unroll
  for (int i = 0; i < 4; ++i) {
    int c = t * 4 + i;
    orow[c] = (bf16_t)((xv[i] - mu) * r * w[c] + b[c]);
  }
}

// ---------------- WMMA GEMM: C[M,N] f32 = A[M,K]bf16 * B[K,N]bf16 ----------
// 128x64 block tile, BK=32, 128 threads = 4 waves.
// Each wave owns 32 rows x 64 cols: 2 A-frags x 4 B-frags -> 8 WMMA / K-step.
__global__ __launch_bounds__(128) void gemm_bf16_kernel(
    const bf16_t* __restrict__ A, const bf16_t* __restrict__ Bm,
    float* __restrict__ C, int M, int N, int K) {
  __shared__ __align__(16) bf16_t As[128 * 32];
  __shared__ __align__(16) bf16_t Bt[64 * 32];   // transposed: Bt[n][k]
  const int tid = threadIdx.x;
  const int wid = tid >> 5;
  const int lane = tid & 31;
  const int l15 = lane & 15;
  const int half = lane >> 4;
  const int m0 = blockIdx.y * 128;
  const int n0 = blockIdx.x * 64;
  f32x8 acc[2][4] = {};

  // hoisted bases; all in-loop offsets are 32-bit
  const bf16_t* gA = A + (size_t)(m0 + tid) * K;
  const int kq = (tid & 7) * 4;                  // k quad base (B transpose)
  const int ns = (tid >> 3) * 4;                 // n strip base
  const bf16_t* gB = Bm + (size_t)kq * N + n0 + ns;

  for (int k0 = 0; k0 < K; k0 += 32) {
    {   // stage A: batch 4x16B loads, then 4 ds_store_b128
      bf16x8 ta[4];
#pragma unroll
      for (int j = 0; j < 4; ++j)
        ta[j] = *(const bf16x8*)(gA + (unsigned)(k0 + j * 8));
#pragma unroll
      for (int j = 0; j < 4; ++j)
        *(bf16x8*)(As + tid * 32 + j * 8) = ta[j];
      if (k0 + 32 < K) __builtin_prefetch(gA + (unsigned)(k0 + 32), 0, 3);
    }
    {   // stage B transposed with 8B-packed writes: thread covers 4n x 4k
      const bf16_t* g = gB + (unsigned)(k0 * N);
      bf16x4 brow[4];
#pragma unroll
      for (int kk = 0; kk < 4; ++kk) brow[kk] = *(const bf16x4*)(g + (unsigned)(kk * N));
#pragma unroll
      for (int j = 0; j < 4; ++j) {
        bf16x4 pk = {brow[0][j], brow[1][j], brow[2][j], brow[3][j]};
        *(bf16x4*)(Bt + (ns + j) * 32 + kq) = pk;
      }
      if (k0 + 32 < K) __builtin_prefetch(g + (unsigned)(32 * N), 0, 3);
    }
    __syncthreads();
    bf16x16 a0 = load_frag(As, 32, wid * 32 + l15, 0);
    bf16x16 a1 = load_frag(As, 32, wid * 32 + 16 + l15, 0);
    bf16x16 bfr[4];
#pragma unroll
    for (int nt = 0; nt < 4; ++nt) bfr[nt] = load_frag(Bt, 32, nt * 16 + l15, 0);
#pragma unroll
    for (int nt = 0; nt < 4; ++nt) acc[0][nt] = wmma_bf16f32(a0, bfr[nt], acc[0][nt]);
#pragma unroll
    for (int nt = 0; nt < 4; ++nt) acc[1][nt] = wmma_bf16f32(a1, bfr[nt], acc[1][nt]);
    __syncthreads();
  }
  // epilogue: 32-bit offsets from one base
  const unsigned cbase = (unsigned)((m0 + wid * 32 + 8 * half) * N + n0 + l15);
#pragma unroll
  for (int am = 0; am < 2; ++am)
#pragma unroll
    for (int nt = 0; nt < 4; ++nt)
#pragma unroll
      for (int r = 0; r < 8; ++r)
        C[cbase + (unsigned)((am * 16 + r) * N + nt * 16)] = acc[am][nt][r];
}

// ---------------- RoPE + layout transform to [B,H,S,HD] bf16 ----------------
__global__ __launch_bounds__(256) void rope_qkv_kernel(
    const float* __restrict__ qf, const float* __restrict__ kf,
    const float* __restrict__ vf, const float* __restrict__ fcos,
    const float* __restrict__ fsin, bf16_t* __restrict__ qb,
    bf16_t* __restrict__ kb, bf16_t* __restrict__ vb) {
  const int tok = blockIdx.x;
  const int b = tok / SS, s = tok % SS;
  const int t = threadIdx.x;
  const float* qr = qf + (size_t)tok * SD;
  const float* kr = kf + (size_t)tok * SKV;
  const float* vr = vf + (size_t)tok * SKV;
  const float* c  = fcos + (size_t)s * (SHD / 2);
  const float* sn = fsin + (size_t)s * (SHD / 2);
#pragma unroll
  for (int ii = 0; ii < 2; ++ii) {          // 512 q rotation pairs
    int i = t + ii * 256;
    int h = i >> 5, j = i & 31;
    float re = qr[h * SHD + 2 * j], im = qr[h * SHD + 2 * j + 1];
    float cc = c[j], ss = sn[j];
    bf16_t* dst = qb + (unsigned)(((b * SH + h) * SS + s) * SHD + 2 * j);
    dst[0] = (bf16_t)(re * cc - im * ss);
    dst[1] = (bf16_t)(re * ss + im * cc);
  }
  if (t < 128) {                            // 128 k rotation pairs
    int h = t >> 5, j = t & 31;
    float re = kr[h * SHD + 2 * j], im = kr[h * SHD + 2 * j + 1];
    float cc = c[j], ss = sn[j];
    bf16_t* dst = kb + (unsigned)(((b * SHKV + h) * SS + s) * SHD + 2 * j);
    dst[0] = (bf16_t)(re * cc - im * ss);
    dst[1] = (bf16_t)(re * ss + im * cc);
  }
  {                                         // 256 v elements straight convert
    int h = t >> 6, d = t & 63;
    vb[(unsigned)(((b * SHKV + h) * SS + s) * SHD + d)] = (bf16_t)vr[t];
  }
}

// ---------------- Flash attention (causal, GQA 4:1) -------------------------
// grid = (S/64, H, B), 128 threads = 4 waves; wave owns 16 q rows.
__global__ __launch_bounds__(128) void flash_attn_kernel(
    const bf16_t* __restrict__ qb, const bf16_t* __restrict__ kb,
    const bf16_t* __restrict__ vb, bf16_t* __restrict__ attn) {
  __shared__ __align__(16) bf16_t Kt[64 * 64];     // [kv][d]
  __shared__ __align__(16) bf16_t Vt[64 * 64];     // transposed [d][kv]
  __shared__ __align__(16) bf16_t Pl[4][16 * 64];  // per-wave P tile [m][kv]
  const int tid = threadIdx.x, wid = tid >> 5, lane = tid & 31;
  const int l15 = lane & 15, half = lane >> 4;
  const int b = blockIdx.z, h = blockIdx.y, q0 = blockIdx.x * 64;
  const int hkv = h / (SH / SHKV);
  const bf16_t* Q  = qb + (size_t)(b * SH + h) * SS * SHD;
  const bf16_t* Kg = kb + (size_t)(b * SHKV + hkv) * SS * SHD;
  const bf16_t* Vg = vb + (size_t)(b * SHKV + hkv) * SS * SHD;

  // Q fragments resident in VGPRs for whole kernel (d split 0..31 / 32..63)
  bf16x16 aq0 = load_frag(Q, SHD, q0 + wid * 16 + l15, 0);
  bf16x16 aq1 = load_frag(Q, SHD, q0 + wid * 16 + l15, 32);

  f32x8 oacc[4] = {};
  float mrow[8], lrow[8];
#pragma unroll
  for (int r = 0; r < 8; ++r) { mrow[r] = -1e30f; lrow[r] = 0.f; }

  // hoisted staging bases (32-bit in-loop offsets)
  const int krow = tid >> 1, kcol = (tid & 1) * 32;      // K staging
  const bf16_t* gK = Kg + (unsigned)(krow * SHD + kcol);
  const int rq = (tid & 15) * 4;                         // V staging: kv quad
  const int ds = (tid >> 4) * 8;                         // V staging: d strip
  const bf16_t* gV = Vg + (unsigned)(rq * SHD + ds);

  const int ntiles = blockIdx.x + 1;   // causal: kv tiles with k0 <= q0+63
  for (int ti = 0; ti < ntiles; ++ti) {
    const int k0 = ti * 64;
    {   // stage K tile: batch 4x16B loads, then 4 ds_store_b128
      const bf16_t* g = gK + (unsigned)(k0 * SHD);
      bf16x8 tk[4];
#pragma unroll
      for (int j = 0; j < 4; ++j) tk[j] = *(const bf16x8*)(g + j * 8);
#pragma unroll
      for (int j = 0; j < 4; ++j)
        *(bf16x8*)(Kt + krow * 64 + kcol + j * 8) = tk[j];
      if (ti + 1 < ntiles) __builtin_prefetch(g + (unsigned)(64 * SHD), 0, 3);
    }
    {   // stage V transposed with 8B-packed writes: thread covers 4kv x 8d
      const bf16_t* g = gV + (unsigned)(k0 * SHD);
      bf16x8 vrow[4];
#pragma unroll
      for (int rr = 0; rr < 4; ++rr) vrow[rr] = *(const bf16x8*)(g + (unsigned)(rr * SHD));
#pragma unroll
      for (int j = 0; j < 8; ++j) {
        bf16x4 pk = {vrow[0][j], vrow[1][j], vrow[2][j], vrow[3][j]};
        *(bf16x4*)(Vt + (ds + j) * 64 + rq) = pk;
      }
      if (ti + 1 < ntiles) __builtin_prefetch(g + (unsigned)(64 * SHD), 0, 3);
    }
    __syncthreads();

    // S = Q * K^T : B-frag (k=d, n=kv) is contiguous in Kt[kv][d]
    f32x8 sacc[4] = {};
    {
      bf16x16 bk[4];
#pragma unroll
      for (int nt = 0; nt < 4; ++nt) bk[nt] = load_frag(Kt, 64, nt * 16 + l15, 0);
#pragma unroll
      for (int nt = 0; nt < 4; ++nt) sacc[nt] = wmma_bf16f32(aq0, bk[nt], sacc[nt]);
#pragma unroll
      for (int nt = 0; nt < 4; ++nt) bk[nt] = load_frag(Kt, 64, nt * 16 + l15, 32);
#pragma unroll
      for (int nt = 0; nt < 4; ++nt) sacc[nt] = wmma_bf16f32(aq1, bk[nt], sacc[nt]);
    }

    // scale + causal mask + running row max (reduce within 16-lane half)
    float rmax[8];
#pragma unroll
    for (int r = 0; r < 8; ++r) {
      const int qrow = q0 + wid * 16 + r + 8 * half;
      float v = -1e30f;
#pragma unroll
      for (int nt = 0; nt < 4; ++nt) {
        int kc = k0 + nt * 16 + l15;
        float sv = sacc[nt][r] * 0.125f;          // 1/sqrt(64)
        if (kc > qrow) sv = -1e30f;
        sacc[nt][r] = sv;
        v = fmaxf(v, sv);
      }
#pragma unroll
      for (int off = 1; off < 16; off <<= 1) v = fmaxf(v, __shfl_xor(v, off, 32));
      rmax[r] = v;
    }

    // online softmax update; emit P (bf16) to LDS for layout conversion
#pragma unroll
    for (int r = 0; r < 8; ++r) {
      float mnew = fmaxf(mrow[r], rmax[r]);
      float esc  = __expf(mrow[r] - mnew);
      mrow[r] = mnew;
      float rs = 0.f;
#pragma unroll
      for (int nt = 0; nt < 4; ++nt) {
        float p = __expf(sacc[nt][r] - mnew);
        rs += p;
        Pl[wid][(r + 8 * half) * 64 + nt * 16 + l15] = (bf16_t)p;
        oacc[nt][r] *= esc;
      }
#pragma unroll
      for (int off = 1; off < 16; off <<= 1) rs += __shfl_xor(rs, off, 32);
      lrow[r] = lrow[r] * esc + rs;
    }
    __syncthreads();

    // O += P * V : A-frag from Pl (row-major), B-frag contiguous in Vt[d][kv]
    {
      bf16x16 ap0 = load_frag(Pl[wid], 64, l15, 0);
      bf16x16 ap1 = load_frag(Pl[wid], 64, l15, 32);
      bf16x16 bv[4];
#pragma unroll
      for (int nt = 0; nt < 4; ++nt) bv[nt] = load_frag(Vt, 64, nt * 16 + l15, 0);
#pragma unroll
      for (int nt = 0; nt < 4; ++nt) oacc[nt] = wmma_bf16f32(ap0, bv[nt], oacc[nt]);
#pragma unroll
      for (int nt = 0; nt < 4; ++nt) bv[nt] = load_frag(Vt, 64, nt * 16 + l15, 32);
#pragma unroll
      for (int nt = 0; nt < 4; ++nt) oacc[nt] = wmma_bf16f32(ap1, bv[nt], oacc[nt]);
    }
    __syncthreads();   // before Kt/Vt overwrite next iteration
  }

  // epilogue: normalize and emit attn output in [B,S,H*HD] bf16 for O-proj
  const unsigned obase =
      (unsigned)(((b * SS + q0 + wid * 16 + 8 * half) * SH + h) * SHD + l15);
#pragma unroll
  for (int nt = 0; nt < 4; ++nt)
#pragma unroll
    for (int r = 0; r < 8; ++r) {
      float o = oacc[nt][r] / lrow[r];
      attn[obase + (unsigned)(r * SH * SHD + nt * 16)] = (bf16_t)o;
    }
}

// ---------------- driver ----------------
extern "C" void kernel_launch(void* const* d_in, const int* in_sizes, int n_in,
                              void* d_out, int out_size, void* d_ws, size_t ws_size,
                              hipStream_t stream) {
  (void)in_sizes; (void)n_in; (void)out_size; (void)ws_size;
  const float* x    = (const float*)d_in[0];
  const float* wq   = (const float*)d_in[1];
  const float* wk   = (const float*)d_in[2];
  const float* wv   = (const float*)d_in[3];
  const float* wo   = (const float*)d_in[4];
  const float* lnw  = (const float*)d_in[5];
  const float* lnb  = (const float*)d_in[6];
  const float* fcos = (const float*)d_in[7];
  const float* fsin = (const float*)d_in[8];
  float* out = (float*)d_out;

  char* ws = (char*)d_ws;
  size_t off = 0;
  auto carve = [&](size_t bytes) -> void* {
    void* p = ws + off;
    off += (bytes + 255) & ~(size_t)255;
    return p;
  };
  bf16_t* xnb   = (bf16_t*)carve((size_t)SM * SD * 2);
  bf16_t* wqb   = (bf16_t*)carve((size_t)SD * SD * 2);
  bf16_t* wkb   = (bf16_t*)carve((size_t)SD * SKV * 2);
  bf16_t* wvb   = (bf16_t*)carve((size_t)SD * SKV * 2);
  bf16_t* wob   = (bf16_t*)carve((size_t)SD * SD * 2);
  float*  qf    = (float*)carve((size_t)SM * SD * 4);
  float*  kf    = (float*)carve((size_t)SM * SKV * 4);
  float*  vf    = (float*)carve((size_t)SM * SKV * 4);
  bf16_t* qrope = (bf16_t*)carve((size_t)SB * SH * SS * SHD * 2);
  bf16_t* krope = (bf16_t*)carve((size_t)SB * SHKV * SS * SHD * 2);
  bf16_t* vrope = (bf16_t*)carve((size_t)SB * SHKV * SS * SHD * 2);
  bf16_t* attnb = (bf16_t*)carve((size_t)SM * SD * 2);

  // 1) weights -> bf16
  cvt_bf16_kernel<<<(SD * SD) / 1024, 256, 0, stream>>>(wq, wqb, SD * SD);
  cvt_bf16_kernel<<<(SD * SKV) / 1024, 256, 0, stream>>>(wk, wkb, SD * SKV);
  cvt_bf16_kernel<<<(SD * SKV) / 1024, 256, 0, stream>>>(wv, wvb, SD * SKV);
  cvt_bf16_kernel<<<(SD * SD) / 1024, 256, 0, stream>>>(wo, wob, SD * SD);

  // 2) layernorm -> bf16 activations
  layernorm_kernel<<<SM, 256, 0, stream>>>(x, lnw, lnb, xnb);

  // 3) QKV projections (WMMA)
  gemm_bf16_kernel<<<dim3(SD / 64, SM / 128), 128, 0, stream>>>(xnb, wqb, qf, SM, SD, SD);
  gemm_bf16_kernel<<<dim3(SKV / 64, SM / 128), 128, 0, stream>>>(xnb, wkb, kf, SM, SKV, SD);
  gemm_bf16_kernel<<<dim3(SKV / 64, SM / 128), 128, 0, stream>>>(xnb, wvb, vf, SM, SKV, SD);

  // 4) RoPE + head-major bf16 layouts
  rope_qkv_kernel<<<SM, 256, 0, stream>>>(qf, kf, vf, fcos, fsin, qrope, krope, vrope);

  // 5) causal GQA flash attention (WMMA)
  flash_attn_kernel<<<dim3(SS / 64, SH, SB), 128, 0, stream>>>(qrope, krope, vrope, attnb);

  // 6) output projection (WMMA) -> f32 d_out
  gemm_bf16_kernel<<<dim3(SD / 64, SM / 128), 128, 0, stream>>>(attnb, wob, out, SM, SD, SD);
}
